// GatedDeltaNetBlock_61641370632598
// MI455X (gfx1250) — compile-verified, matrix operand order checked by reading, not run on hardware
//
#include <hip/hip_runtime.h>
#include <hip/hip_bf16.h>
#include <stdint.h>

// ---- problem dims ----
#define BB   4
#define TT   2048
#define BTm  (BB * TT)          // 8192 rows
#define DDm  1024
#define HHm  16
#define DKm  48
#define DVm  96
#define HDKm (HHm * DKm)        // 768
#define HDVm (HHm * DVm)        // 1536
#define KWm  4
#define IIm  3072
#define NCAT  4640              // q|k|v|gate|b|a columns
#define NCATP 4736              // padded to 37 * 128

typedef __bf16 bf16_t;
typedef __bf16 v16bf __attribute__((ext_vector_type(16)));
typedef float  v8f   __attribute__((ext_vector_type(8)));
typedef unsigned int u32x4 __attribute__((ext_vector_type(4)));
typedef int          i32x8 __attribute__((ext_vector_type(8)));
typedef int          i32x4 __attribute__((ext_vector_type(4)));

#if __has_builtin(__builtin_amdgcn_tensor_load_to_lds)
#define HAVE_TDM 1
#else
#define HAVE_TDM 0
#endif

static __device__ __forceinline__ bf16_t f2bf(float f) {
  unsigned u = __builtin_bit_cast(unsigned, f);
  unsigned r = u + 0x7FFFu + ((u >> 16) & 1u);   // round-to-nearest-even
  unsigned short h = (unsigned short)(r >> 16);
  return __builtin_bit_cast(bf16_t, h);
}
static __device__ __forceinline__ float sigmoidf_(float x) { return 1.f / (1.f + __expf(-x)); }
static __device__ __forceinline__ float siluf_(float x)    { return x * sigmoidf_(x); }

static __device__ __forceinline__ void tensor_wait0() {
#if __has_builtin(__builtin_amdgcn_s_wait_tensorcnt)
  __builtin_amdgcn_s_wait_tensorcnt(0);
#else
  asm volatile("s_wait_tensorcnt 0x0" ::: "memory");
#endif
}

#if HAVE_TDM
// Issue one TDM 2D tile load: global (strided rows) -> LDS (packed rows).
// D# built per CDNA5 ISA 8.3-8.5: data_size=2B, tile_dim0 x tile_dim1,
// tensor_dim0_stride in elements; groups 2/3 zero (2D tensor).
// Toolchain uses the 6-operand builtin: (g0, g1, g2, g3, extra, cpol).
static __device__ __forceinline__ void tdm_load_2d(
    unsigned lds_addr, const bf16_t* gptr,
    unsigned tensor_d0, unsigned tensor_d1,
    unsigned tile_d0, unsigned tile_d1,
    unsigned long long stride0) {
  unsigned long long ga = (unsigned long long)(uintptr_t)gptr;
  u32x4 g0;
  g0[0] = 1u;                                            // count=1, user descriptor
  g0[1] = lds_addr;                                      // LDS byte address
  g0[2] = (unsigned)(ga & 0xFFFFFFFFu);                  // global_addr[31:0]
  g0[3] = (unsigned)((ga >> 32) & 0x01FFFFFFu) | (2u << 30);  // ga[56:32] | type=2
  i32x8 g1;
  g1[0] = (int)(1u << 16);                               // wg_mask=0, data_size=1 (2B)
  g1[1] = (int)((tensor_d0 & 0xFFFFu) << 16);            // atomic_addr=0 | td0[15:0]
  g1[2] = (int)((tensor_d0 >> 16) | ((tensor_d1 & 0xFFFFu) << 16));
  g1[3] = (int)((tensor_d1 >> 16) | (tile_d0 << 16));    // td1[31:16] | tile_dim0
  g1[4] = (int)(tile_d1 & 0xFFFFu);                      // tile_dim1 | tile_dim2=0
  g1[5] = (int)(unsigned)(stride0 & 0xFFFFFFFFu);        // dim0_stride[31:0]
  g1[6] = (int)(unsigned)((stride0 >> 32) & 0xFFFFu);    // dim0_stride[47:32] | dim1_stride lo=0
  g1[7] = 0;
  i32x4 gz4 = {0, 0, 0, 0};
  i32x8 gz8 = {0, 0, 0, 0, 0, 0, 0, 0};
  __builtin_amdgcn_tensor_load_to_lds(g0, g1, gz4, gz4, gz8, 0);
}
#endif

// ============================================================
// WMMA GEMM: C[M,N] = A[M,K](bf16,row-major) * Bt[N,K](bf16,K-major)^T (+Res)
// Macro tile 128x128, BK=32, 256 threads = 8 waves (4 M x 2 N),
// wave tile 32x64 -> 2x4 v_wmma_f32_16x16x32_bf16 accumulators.
// Tiles staged to LDS by the Tensor Data Mover (wave 0 issues 2 TDM
// descriptors per k-step, double buffered, s_wait_tensorcnt before barrier).
// ============================================================
template <int EPI>
__global__ __launch_bounds__(256, 2)
void wmma_gemm_kernel(const bf16_t* __restrict__ A, const bf16_t* __restrict__ Bt,
                      const float* __restrict__ Res, float* __restrict__ C,
                      int M, int N, int Kd) {
  __shared__ alignas(32) bf16_t Asm[2][128][32];   // [m][k]
  __shared__ alignas(32) bf16_t Bsm[2][128][32];   // [n][k]

  const int tid  = threadIdx.x;
  const int lane = tid & 31;
  const int wave = tid >> 5;
  const int bm0  = blockIdx.y * 128;
  const int bn0  = blockIdx.x * 128;
  const int waveM = wave & 3;    // 4 waves along M, 32 rows each
  const int waveN = wave >> 2;   // 2 waves along N, 64 cols each

  v8f acc[2][4] = {};
  const int KT = Kd >> 5;
  int buf = 0;

#if !HAVE_TDM
  // fallback staging: thread covers 16 contiguous K of one row of each tile
  const int arow  = tid >> 1;
  const int ahalf = tid & 1;
  auto stage_fb = [&](int kt, int bsel) {
    const int k0 = kt << 5;
    const uint4* ag = (const uint4*)(A + (size_t)(bm0 + arow) * Kd + k0 + ahalf * 16);
    uint4 a0 = ag[0], a1 = ag[1];
    const uint4* bg = (const uint4*)(Bt + (size_t)(bn0 + arow) * Kd + k0 + ahalf * 16);
    uint4 b0 = bg[0], b1 = bg[1];
    uint4* as = (uint4*)&Asm[bsel][arow][ahalf * 16];
    as[0] = a0; as[1] = a1;
    uint4* bs = (uint4*)&Bsm[bsel][arow][ahalf * 16];
    bs[0] = b0; bs[1] = b1;
  };
#endif

#if HAVE_TDM
  if (tid < 32) {   // wave 0 drives the TDM
    tdm_load_2d((unsigned)(uintptr_t)(void*)&Asm[0][0][0], A + (size_t)bm0 * Kd,
                (unsigned)Kd, (unsigned)M, 32u, 128u, (unsigned long long)Kd);
    tdm_load_2d((unsigned)(uintptr_t)(void*)&Bsm[0][0][0], Bt + (size_t)bn0 * Kd,
                (unsigned)Kd, (unsigned)N, 32u, 128u, (unsigned long long)Kd);
    tensor_wait0();
  }
#else
  stage_fb(0, 0);
#endif
  __syncthreads();

  const int row16 = lane & 15;
  const int hlf   = lane >> 4;

  for (int kt = 0; kt < KT; ++kt) {
    const bool more = (kt + 1 < KT);
#if HAVE_TDM
    if (more && tid < 32) {
      const int k0n = (kt + 1) << 5;
      tdm_load_2d((unsigned)(uintptr_t)(void*)&Asm[buf ^ 1][0][0],
                  A + (size_t)bm0 * Kd + k0n, (unsigned)Kd, (unsigned)M, 32u, 128u,
                  (unsigned long long)Kd);
      tdm_load_2d((unsigned)(uintptr_t)(void*)&Bsm[buf ^ 1][0][0],
                  Bt + (size_t)bn0 * Kd + k0n, (unsigned)Kd, (unsigned)N, 32u, 128u,
                  (unsigned long long)Kd);
    }
#else
    if (more) {
      __builtin_prefetch(A + (size_t)(bm0 + arow) * Kd + ((kt + 1) << 5), 0, 1);
      stage_fb(kt + 1, buf ^ 1);
    }
#endif
    v16bf afrag[2], bfrag[4];
#pragma unroll
    for (int mi = 0; mi < 2; ++mi)
      afrag[mi] = *(const v16bf*)&Asm[buf][waveM * 32 + mi * 16 + row16][hlf * 16];
#pragma unroll
    for (int ni = 0; ni < 4; ++ni)
      bfrag[ni] = *(const v16bf*)&Bsm[buf][waveN * 64 + ni * 16 + row16][hlf * 16];
#pragma unroll
    for (int mi = 0; mi < 2; ++mi)
#pragma unroll
      for (int ni = 0; ni < 4; ++ni)
        acc[mi][ni] = __builtin_amdgcn_wmma_f32_16x16x32_bf16(
            false, afrag[mi], false, bfrag[ni], (short)0, acc[mi][ni], false, false);
#if HAVE_TDM
    if (more && tid < 32) tensor_wait0();
#endif
    __syncthreads();
    buf ^= 1;
  }

  // epilogue per ISA C/D layout: vgpr i -> row (hlf*8 + i), lane%16 -> col
#pragma unroll
  for (int mi = 0; mi < 2; ++mi) {
    const int rbase = bm0 + waveM * 32 + mi * 16 + hlf * 8;
#pragma unroll
    for (int ni = 0; ni < 4; ++ni) {
      const int cn = bn0 + waveN * 64 + ni * 16 + row16;
#pragma unroll
      for (int i = 0; i < 8; ++i) {
        const size_t idx = (size_t)(rbase + i) * N + cn;
        float v = acc[mi][ni][i];
        if (EPI) v += Res[idx];
        C[idx] = v;
      }
    }
  }
}

// ---- RMSNorm (f32 in) -> bf16 out, one block per row of D=1024 ----
__global__ __launch_bounds__(256)
void rmsnorm_bf16_kernel(const float* __restrict__ x, const float* __restrict__ w,
                         bf16_t* __restrict__ out) {
  const int row = blockIdx.x;
  const float* xr = x + (size_t)row * DDm;
  const int tid = threadIdx.x, lane = tid & 31, wv = tid >> 5;
  float ss = 0.f;
  for (int i = tid; i < DDm; i += 256) { float v = xr[i]; ss += v * v; }
#pragma unroll
  for (int off = 16; off; off >>= 1) ss += __shfl_xor(ss, off, 32);
  __shared__ float red[8];
  if (lane == 0) red[wv] = ss;
  __syncthreads();
  float tot = 0.f;
#pragma unroll
  for (int i = 0; i < 8; ++i) tot += red[i];
  const float sc = rsqrtf(tot * (1.f / DDm) + 1e-5f);
  bf16_t* orow = out + (size_t)row * DDm;
  for (int i = tid; i < DDm; i += 256) orow[i] = f2bf(xr[i] * sc * w[i]);
}

// ---- f32 [K x N] row-major -> bf16 TRANSPOSED: dst[(rowOff+n)*dstStride + k] ----
__global__ void cvt_matT_bf16_kernel(const float* __restrict__ src, bf16_t* __restrict__ dst,
                                     int rowsK, int colsN, int dstStride, int rowOff) {
  size_t idx = (size_t)blockIdx.x * 256 + threadIdx.x;
  size_t total = (size_t)rowsK * colsN;
  if (idx >= total) return;
  int k = (int)(idx / colsN), n = (int)(idx % colsN);
  dst[(size_t)(rowOff + n) * dstStride + k] = f2bf(src[idx]);
}

__global__ void zero_bf16_kernel(bf16_t* __restrict__ dst, size_t n) {
  size_t idx = (size_t)blockIdx.x * 256 + threadIdx.x;
  if (idx < n) dst[idx] = __builtin_bit_cast(bf16_t, (unsigned short)0);
}

// ---- causal depthwise conv (K=4) + SiLU over q|k|v channels ----
__global__ void conv_silu_kernel(const float* __restrict__ pre, const float* __restrict__ wq,
                                 const float* __restrict__ wk, const float* __restrict__ wv,
                                 float* __restrict__ qkv) {
  size_t idx = (size_t)blockIdx.x * 256 + threadIdx.x;
  if (idx >= (size_t)BTm * 3072) return;
  const int c = (int)(idx % 3072);
  const long bt = (long)(idx / 3072);
  const int tl = (int)(bt % TT);
  const float* w4 = (c < HDKm)     ? (wq + c * KWm)
                  : (c < 2 * HDKm) ? (wk + (c - HDKm) * KWm)
                                   : (wv + (c - 2 * HDKm) * KWm);
  float acc = 0.f;
#pragma unroll
  for (int j = 0; j < KWm; ++j) {
    int tt = tl - (KWm - 1) + j;
    if (tt >= 0) acc += w4[j] * pre[(size_t)(bt - (KWm - 1) + j) * NCATP + c];
  }
  qkv[idx] = siluf_(acc);
}

// ---- L2 norm over DK for q and k, in place ----
__global__ void l2norm_qk_kernel(float* __restrict__ qkv) {
  size_t idx = (size_t)blockIdx.x * 256 + threadIdx.x;
  if (idx >= (size_t)BTm * HHm * 2) return;
  const int h = (int)(idx % HHm);
  size_t rest = idx / HHm;
  const int s = (int)(rest % 2);
  const size_t bt = rest / 2;
  float* p = qkv + bt * 3072 + s * HDKm + h * DKm;
  float ss = 0.f;
#pragma unroll
  for (int i = 0; i < DKm; ++i) { float v = p[i]; ss += v * v; }
  const float sc = rsqrtf(ss + 1e-6f);
#pragma unroll
  for (int i = 0; i < DKm; ++i) p[i] *= sc;
}

// ---- mamba gate g and beta ----
__global__ void gate_beta_kernel(const float* __restrict__ pre, const float* __restrict__ A_log,
                                 const float* __restrict__ dt_bias, float* __restrict__ gb) {
  size_t idx = (size_t)blockIdx.x * 256 + threadIdx.x;
  if (idx >= (size_t)BTm * HHm) return;
  const int h = (int)(idx % HHm);
  const size_t bt = idx / HHm;
  const float braw = pre[bt * NCATP + 3072 + HDVm + h];
  const float araw = pre[bt * NCATP + 3072 + HDVm + HHm + h];
  const float beta = sigmoidf_(braw);
  const float xa = araw + dt_bias[h];
  const float sp = (xa > 20.f) ? xa : log1pf(__expf(xa));
  gb[bt * 32 + h]      = -__expf(A_log[h]) * sp;
  gb[bt * 32 + 16 + h] = beta;
}

// ---- gated delta rule scan: one block per (b,h), 96 lanes = one dv each,
// state column S[48] in VGPRs, q/k broadcast via LDS per step ----
__global__ __launch_bounds__(96)
void delta_scan_kernel(const float* __restrict__ qkv, const float* __restrict__ gb,
                       float* __restrict__ o) {
  const int bh = blockIdx.x;
  const int b = bh >> 4, h = bh & 15;
  const int dv = threadIdx.x;
  __shared__ float qs[DKm], ks[DKm];
  float S[DKm];
#pragma unroll
  for (int i = 0; i < DKm; ++i) S[i] = 0.f;
  const float scale = 0.14433756729740643f;  // 48^-0.5
  for (int t = 0; t < TT; ++t) {
    const size_t row = (size_t)b * TT + t;
    const float* base = qkv + row * 3072;
    if (dv < DKm) qs[dv] = base[h * DKm + dv] * scale;
    else          ks[dv - DKm] = base[HDKm + h * DKm + (dv - DKm)];
    __syncthreads();
    const float g   = gb[row * 32 + h];
    const float bet = gb[row * 32 + 16 + h];
    const float eg  = __expf(g);
    const float v   = base[2 * HDKm + h * DVm + dv];
    float kS = 0.f;
#pragma unroll
    for (int i = 0; i < DKm; ++i) kS += ks[i] * S[i];
    const float vnew = (v - kS * eg) * bet;
    float ot = 0.f;
#pragma unroll
    for (int i = 0; i < DKm; ++i) {
      float s2 = S[i] * eg + ks[i] * vnew;
      S[i] = s2;
      ot += qs[i] * s2;
    }
    o[row * HDVm + h * DVm + dv] = ot;
    __syncthreads();
  }
}

// ---- gated RMSNorm over DV, times silu(gate), -> bf16 for o-proj ----
__global__ void gated_rmsnorm_kernel(const float* __restrict__ o, const float* __restrict__ pre,
                                     const float* __restrict__ onw, bf16_t* __restrict__ obf) {
  size_t idx = (size_t)blockIdx.x * 256 + threadIdx.x;
  if (idx >= (size_t)BTm * HHm) return;
  const int h = (int)(idx % HHm);
  const size_t bt = idx / HHm;
  const float* orow = o + bt * HDVm + h * DVm;
  const float* grow = pre + bt * NCATP + 3072 + h * DVm;
  float ss = 0.f;
#pragma unroll 8
  for (int i = 0; i < DVm; ++i) { float v = orow[i]; ss += v * v; }
  const float sc = rsqrtf(ss * (1.f / DVm) + 1e-5f);
#pragma unroll 8
  for (int i = 0; i < DVm; ++i) {
    float gate = grow[i];
    obf[bt * HDVm + h * DVm + i] = f2bf(orow[i] * sc * onw[i] * siluf_(gate));
  }
}

// ---- swiglu: silu(gate) * up -> bf16 ----
__global__ void swiglu_kernel(const float* __restrict__ gu, bf16_t* __restrict__ out) {
  size_t idx = (size_t)blockIdx.x * 256 + threadIdx.x;
  if (idx >= (size_t)BTm * IIm) return;
  const int c = (int)(idx % IIm);
  const size_t bt = idx / IIm;
  const float gp = gu[bt * 2 * IIm + c];
  const float up = gu[bt * 2 * IIm + IIm + c];
  out[idx] = f2bf(siluf_(gp) * up);
}

extern "C" void kernel_launch(void* const* d_in, const int* in_sizes, int n_in,
                              void* d_out, int out_size, void* d_ws, size_t ws_size,
                              hipStream_t stream) {
  (void)in_sizes; (void)n_in; (void)out_size; (void)ws_size;
  const float* x      = (const float*)d_in[0];
  const float* norm1w = (const float*)d_in[1];
  const float* q_w    = (const float*)d_in[2];
  const float* k_w    = (const float*)d_in[3];
  const float* v_w    = (const float*)d_in[4];
  const float* cq_w   = (const float*)d_in[5];
  const float* ck_w   = (const float*)d_in[6];
  const float* cv_w   = (const float*)d_in[7];
  const float* b_w    = (const float*)d_in[8];
  const float* a_w    = (const float*)d_in[9];
  const float* A_log  = (const float*)d_in[10];
  const float* dt_b   = (const float*)d_in[11];
  const float* g_w    = (const float*)d_in[12];
  const float* onw    = (const float*)d_in[13];
  const float* o_w    = (const float*)d_in[14];
  const float* norm2w = (const float*)d_in[15];
  const float* gate_w = (const float*)d_in[16];
  const float* up_w   = (const float*)d_in[17];
  const float* down_w = (const float*)d_in[18];
  float* out = (float*)d_out;

  char* ws = (char*)d_ws;
  size_t off = 0;
  auto alloc = [&](size_t bytes) -> char* {
    char* p = ws + off;
    off += (bytes + 255) & ~(size_t)255;
    return p;
  };
  bf16_t* h_bf  = (bf16_t*)alloc((size_t)BTm * DDm * 2);
  bf16_t* WcatT = (bf16_t*)alloc((size_t)NCATP * DDm * 2);   // [NCATP][1024]
  float*  pre   = (float*) alloc((size_t)BTm * NCATP * 4);
  float*  qkv   = (float*) alloc((size_t)BTm * 3072 * 4);
  float*  gb    = (float*) alloc((size_t)BTm * 32 * 4);
  float*  obuf  = (float*) alloc((size_t)BTm * HDVm * 4);
  bf16_t* o_bf  = (bf16_t*)alloc((size_t)BTm * HDVm * 2);
  bf16_t* WoT   = (bf16_t*)alloc((size_t)DDm * HDVm * 2);    // [1024][1536]
  float*  x1    = (float*) alloc((size_t)BTm * DDm * 4);
  bf16_t* h2_bf = (bf16_t*)alloc((size_t)BTm * DDm * 2);
  bf16_t* WguT  = (bf16_t*)alloc((size_t)2 * IIm * DDm * 2); // [6144][1024]
  float*  gupre = (float*) alloc((size_t)BTm * 2 * IIm * 4);
  bf16_t* gu_bf = (bf16_t*)alloc((size_t)BTm * IIm * 2);
  bf16_t* WdT   = (bf16_t*)alloc((size_t)DDm * IIm * 2);     // [1024][3072]

  auto cdiv = [](size_t a, size_t b) { return (unsigned)((a + b - 1) / b); };

  // weights -> bf16, transposed to [N][K] (pad rows of WcatT zeroed first)
  zero_bf16_kernel<<<cdiv((size_t)NCATP * DDm, 256), 256, 0, stream>>>(WcatT, (size_t)NCATP * DDm);
  cvt_matT_bf16_kernel<<<cdiv((size_t)DDm * HDKm, 256), 256, 0, stream>>>(q_w, WcatT, DDm, HDKm, DDm, 0);
  cvt_matT_bf16_kernel<<<cdiv((size_t)DDm * HDKm, 256), 256, 0, stream>>>(k_w, WcatT, DDm, HDKm, DDm, HDKm);
  cvt_matT_bf16_kernel<<<cdiv((size_t)DDm * HDVm, 256), 256, 0, stream>>>(v_w, WcatT, DDm, HDVm, DDm, 2 * HDKm);
  cvt_matT_bf16_kernel<<<cdiv((size_t)DDm * HDVm, 256), 256, 0, stream>>>(g_w, WcatT, DDm, HDVm, DDm, 3072);
  cvt_matT_bf16_kernel<<<cdiv((size_t)DDm * HHm, 256), 256, 0, stream>>>(b_w, WcatT, DDm, HHm, DDm, 4608);
  cvt_matT_bf16_kernel<<<cdiv((size_t)DDm * HHm, 256), 256, 0, stream>>>(a_w, WcatT, DDm, HHm, DDm, 4624);
  cvt_matT_bf16_kernel<<<cdiv((size_t)HDVm * DDm, 256), 256, 0, stream>>>(o_w, WoT, HDVm, DDm, HDVm, 0);
  cvt_matT_bf16_kernel<<<cdiv((size_t)DDm * IIm, 256), 256, 0, stream>>>(gate_w, WguT, DDm, IIm, DDm, 0);
  cvt_matT_bf16_kernel<<<cdiv((size_t)DDm * IIm, 256), 256, 0, stream>>>(up_w, WguT, DDm, IIm, DDm, IIm);
  cvt_matT_bf16_kernel<<<cdiv((size_t)IIm * DDm, 256), 256, 0, stream>>>(down_w, WdT, IIm, DDm, IIm, 0);

  // 1) h = rmsnorm(x) -> bf16
  rmsnorm_bf16_kernel<<<BTm, 256, 0, stream>>>(x, norm1w, h_bf);
  // 2) pre = h @ [q|k|v|gate|b|a]   (WMMA + TDM)
  wmma_gemm_kernel<0><<<dim3(NCATP / 128, BTm / 128), 256, 0, stream>>>(h_bf, WcatT, nullptr, pre, BTm, NCATP, DDm);
  // 3) conv + silu -> qkv ; l2norm q,k ; g/beta
  conv_silu_kernel<<<cdiv((size_t)BTm * 3072, 256), 256, 0, stream>>>(pre, cq_w, ck_w, cv_w, qkv);
  l2norm_qk_kernel<<<cdiv((size_t)BTm * HHm * 2, 256), 256, 0, stream>>>(qkv);
  gate_beta_kernel<<<cdiv((size_t)BTm * HHm, 256), 256, 0, stream>>>(pre, A_log, dt_b, gb);
  // 4) gated delta rule sequential scan
  delta_scan_kernel<<<BB * HHm, 96, 0, stream>>>(qkv, gb, obuf);
  // 5) gated rmsnorm -> bf16 ; o-proj + residual (WMMA + TDM)
  gated_rmsnorm_kernel<<<cdiv((size_t)BTm * HHm, 256), 256, 0, stream>>>(obuf, pre, onw, o_bf);
  wmma_gemm_kernel<1><<<dim3(DDm / 128, BTm / 128), 256, 0, stream>>>(o_bf, WoT, x, x1, BTm, DDm, HDVm);
  // 6) MLP: rmsnorm -> gate|up GEMM -> swiglu -> down GEMM + residual
  rmsnorm_bf16_kernel<<<BTm, 256, 0, stream>>>(x1, norm2w, h2_bf);
  wmma_gemm_kernel<0><<<dim3(2 * IIm / 128, BTm / 128), 256, 0, stream>>>(h2_bf, WguT, nullptr, gupre, BTm, 2 * IIm, DDm);
  swiglu_kernel<<<cdiv((size_t)BTm * IIm, 256), 256, 0, stream>>>(gupre, gu_bf);
  wmma_gemm_kernel<1><<<dim3(DDm / 128, BTm / 128), 256, 0, stream>>>(gu_bf, WdT, x1, out, BTm, DDm, IIm);
}
